// BivariateNormalAttention_28097676050869
// MI455X (gfx1250) — compile-verified, matrix operand order checked by reading, not run on hardware
//
#include <hip/hip_runtime.h>
#include <hip/hip_bf16.h>

// ---------------------------------------------------------------------------
// BivariateNormalAttention — MI455X (gfx1250) implementation.
// conv1/conv2 (99.9% of FLOPs) -> implicit-GEMM with v_wmma_f32_16x16x32_bf16.
// Weight tiles moved by the Tensor Data Mover (tensor_load_to_lds, TENSORcnt).
// ---------------------------------------------------------------------------

typedef __bf16 bf16;
typedef __attribute__((ext_vector_type(16))) __bf16 bf16x16;
typedef __attribute__((ext_vector_type(8)))  __bf16 bf16x8;
typedef __attribute__((ext_vector_type(8)))  float  f32x8;
typedef __attribute__((ext_vector_type(4)))  unsigned int u32x4;
typedef __attribute__((ext_vector_type(8)))  int i32x8;
typedef __attribute__((ext_vector_type(4)))  int i32x4;

#define BN_EPS 1e-5f
#define HDIM 112
#define HW   (112*112)   // 12544
#define BATCH 16

// ---------------------------------------------------------------------------
// flat f32 -> bf16 convert (weights)
// ---------------------------------------------------------------------------
__global__ void k_cvt_bf16(const float* __restrict__ in, bf16* __restrict__ out, int n) {
    int i = blockIdx.x * blockDim.x + threadIdx.x;
    if (i < n) out[i] = (bf16)in[i];
}

// ---------------------------------------------------------------------------
// TDM: load one 16-row x 32-col bf16 tile (row stride K elems) into LDS with
// an 80-byte LDS row pitch (64B data + 16B pad via TDM pad feature).
// D# bit layout per CDNA5 ISA §8.3/8.4. One call per wave; EXEC is ignored.
// This toolchain exposes the 6-arg builtin (extra int32x8 descriptor group).
// ---------------------------------------------------------------------------
__device__ __forceinline__ void tdm_load_wtile(unsigned int lds_byte, unsigned long long gaddr,
                                               int K) {
    // wave-uniform scalars for the SGPR descriptor
    unsigned int lds_u = (unsigned int)__builtin_amdgcn_readfirstlane((int)lds_byte);
    unsigned int glo   = (unsigned int)__builtin_amdgcn_readfirstlane((int)(unsigned int)gaddr);
    unsigned int ghi   = (unsigned int)__builtin_amdgcn_readfirstlane((int)(unsigned int)(gaddr >> 32));

    u32x4 g0;
    g0[0] = 1u;                                        // count=1, user descriptor
    g0[1] = lds_u;                                     // lds_addr (bytes)
    g0[2] = glo;                                       // global_addr[31:0]
    g0[3] = (ghi & 0x01FFFFFFu) | (2u << 30);          // global_addr[56:32] | type=2
    i32x8 g1;
    g1[0] = (1 << 16)        // data_size = 2 bytes
          | (1 << 20)        // pad_enable
          | (3 << 22)        // pad_interval: 16 DWORDs (=64B) between pads
          | (3 << 25);       // pad_amount:   4 DWORDs (=16B) pad -> 80B row pitch
    g1[1] = (32 << 16);      // atomic_barrier_addr=0 | tensor_dim0 = 32
    g1[2] = (16 << 16);      // tensor_dim0 hi=0      | tensor_dim1 = 16
    g1[3] = (32 << 16);      // tensor_dim1 hi=0      | tile_dim0 = 32
    g1[4] = 16;              // tile_dim1 = 16 | tile_dim2 = 0
    g1[5] = K;               // tensor_dim0_stride[31:0] (elements)
    g1[6] = 0;               // stride hi | tensor_dim1_stride lo
    g1[7] = 0;
    i32x4 z4 = {0, 0, 0, 0};
    i32x8 z8 = {0, 0, 0, 0, 0, 0, 0, 0};
    __builtin_amdgcn_tensor_load_to_lds(g0, g1, z4, z4, z8, 0);
}

// ---------------------------------------------------------------------------
// Implicit-GEMM 3x3 conv (stride 1, pad 1) + BN + ReLU, bf16 WMMA, f32 acc.
//   in : [16, IC, 112, 112]  (f32 if IN_F32 else bf16)
//   wgt: [OC, IC*9] bf16  (OIHW flat == GEMM-K order)
//   out: [16, OC, 112, 112] bf16
// Block: 256 thr (8 waves). Block tile 128(oc) x 128(pix). Wave tile 64x32.
// A tile staged by TDM (one 16x32 slice per wave); B tile im2col-gathered.
// ---------------------------------------------------------------------------
template<int IC, bool IN_F32>
__global__ __launch_bounds__(256)
void k_conv3x3_bn_relu_wmma(const void* __restrict__ in_v,
                            const bf16* __restrict__ wgt,
                            const float* __restrict__ gamma,
                            const float* __restrict__ beta,
                            bf16* __restrict__ out, int OC)
{
    constexpr int K = IC * 9;
    const int n_blk = blockIdx.x * 128;   // pixel tile base
    const int m_blk = blockIdx.y * 128;   // oc tile base

    __shared__ __align__(16) bf16 As[128][40];   // [oc_local][k_local], 80B pitch
    __shared__ __align__(16) bf16 Bs[128][40];   // [pix_local][k_local]

    const int tid  = threadIdx.x;
    const int lane = tid & 31;
    const int wave = tid >> 5;
    const int wm   = wave & 1;            // 0..1 -> m offset 0/64
    const int wn   = wave >> 1;           // 0..3 -> n offset 0/32/64/96

    // ---- B (im2col) loader mapping: thread -> one k row, 16 consecutive pixels
    const int kk   = tid >> 3;            // 0..31  k_local
    const int pb   = (tid & 7) * 16;      // pixel_local base
    const int p0   = n_blk + pb;
    const int bimg = p0 / HW;             // 128-tile never crosses image (12544%128==0)
    const int hw0  = p0 - bimg * HW;
    const int oy0  = hw0 / HDIM;
    const int ox0  = hw0 - oy0 * HDIM;

    // ---- A tile: per-wave TDM slice (16 oc rows starting at wave*16)
    const unsigned int as_base = (unsigned int)(unsigned long long)(uintptr_t)&As[0][0];
    const unsigned int as_lds  = as_base + (unsigned int)(wave * 16 * 80);
    const unsigned long long wg_base =
        (unsigned long long)(uintptr_t)(wgt + (size_t)(m_blk + wave * 16) * K);

    const float* inF = (const float*)in_v;
    const bf16*  inB = (const bf16*)in_v;

    f32x8 acc[4][2];
    #pragma unroll
    for (int mi = 0; mi < 4; ++mi)
        #pragma unroll
        for (int ni = 0; ni < 2; ++ni)
            acc[mi][ni] = (f32x8){0.f,0.f,0.f,0.f,0.f,0.f,0.f,0.f};

    const int fr   = lane & 15;           // row/col within 16x16 fragment
    const int koff = (lane >> 4) * 8;     // 0 (lanes 0-15) / 8 (lanes 16-31)

    for (int k0 = 0; k0 < K; k0 += 32) {
        // ---- A tile via Tensor Data Mover (one instr per wave, EXEC-ignored)
        tdm_load_wtile(as_lds, wg_base + (unsigned long long)(k0 * 2), K);

        // ---- stage B tile: 32 k x 128 pixels (im2col gather, pad=1, branchless)
        {
            const int k  = k0 + kk;
            const int ic = k / 9;
            const int r  = k - ic * 9;
            const int ky = r / 3;
            const int kx = r - ky * 3;
            const int chan_base = (bimg * IC + ic) * HW;
            int oy = oy0, ox = ox0;
            #pragma unroll
            for (int i = 0; i < 16; ++i) {
                const int iy = oy + ky - 1;
                const int ix = ox + kx - 1;
                const bool ok = ((unsigned)iy < (unsigned)HDIM) & ((unsigned)ix < (unsigned)HDIM);
                const int off = ok ? (chan_base + iy * HDIM + ix) : chan_base;
                float v = IN_F32 ? inF[off] : (float)inB[off];
                v = ok ? v : 0.f;
                Bs[pb + i][kk] = (bf16)v;
                if (++ox == HDIM) { ox = 0; ++oy; }
            }
        }

        __builtin_amdgcn_s_wait_tensorcnt(0);   // own wave's TDM slice done
        __syncthreads();                         // all slices + B stores visible

        // ---- fragments: lanes 0-15 -> K[0..7],K[16..23]; lanes 16-31 -> K[8..15],K[24..31]
        bf16x16 afr[4], bfr[2];
        #pragma unroll
        for (int mi = 0; mi < 4; ++mi) {
            const bf16* pa = &As[wm * 64 + mi * 16 + fr][0];
            bf16x8 lo = *(const bf16x8*)(pa + koff);
            bf16x8 hi = *(const bf16x8*)(pa + 16 + koff);
            #pragma unroll
            for (int e = 0; e < 8; ++e) { afr[mi][e] = lo[e]; afr[mi][8 + e] = hi[e]; }
        }
        #pragma unroll
        for (int ni = 0; ni < 2; ++ni) {
            const bf16* pbr = &Bs[wn * 32 + ni * 16 + fr][0];
            bf16x8 lo = *(const bf16x8*)(pbr + koff);
            bf16x8 hi = *(const bf16x8*)(pbr + 16 + koff);
            #pragma unroll
            for (int e = 0; e < 8; ++e) { bfr[ni][e] = lo[e]; bfr[ni][8 + e] = hi[e]; }
        }

        #pragma unroll
        for (int mi = 0; mi < 4; ++mi)
            #pragma unroll
            for (int ni = 0; ni < 2; ++ni)
                acc[mi][ni] = __builtin_amdgcn_wmma_f32_16x16x32_bf16(
                    false, afr[mi], false, bfr[ni], (short)0, acc[mi][ni], false, false);

        __syncthreads();
    }

    // ---- epilogue: BN (running stats 0/1) + ReLU, write bf16
    // D layout: VGPR vr, lane -> M = vr + (lane>=16)*8, N = lane&15
    const int half8 = (lane >> 4) * 8;
    const float rs = rsqrtf(1.f + BN_EPS);
    #pragma unroll
    for (int ni = 0; ni < 2; ++ni) {
        const int p  = n_blk + wn * 32 + ni * 16 + fr;
        const int b  = p / HW;
        const int hw = p - b * HW;
        #pragma unroll
        for (int mi = 0; mi < 4; ++mi) {
            #pragma unroll
            for (int vr = 0; vr < 8; ++vr) {
                const int oc = m_blk + wm * 64 + mi * 16 + half8 + vr;
                float v = acc[mi][ni][vr] * (gamma[oc] * rs) + beta[oc];
                v = fmaxf(v, 0.f);
                out[(b * OC + oc) * HW + hw] = (bf16)v;
            }
        }
    }
}

// ---------------------------------------------------------------------------
// AdaptiveAvgPool 112 -> 7 (exact 16x16 means). in bf16, out f32.
// ---------------------------------------------------------------------------
__global__ void k_pool16(const bf16* __restrict__ in, float* __restrict__ out) {
    int idx = blockIdx.x * blockDim.x + threadIdx.x;    // 16*256*49
    if (idx >= BATCH * 256 * 49) return;
    int c7 = idx % 49; int bc = idx / 49;
    int py = c7 / 7, px = c7 % 7;
    const bf16* src = in + bc * HW + (py * 16) * HDIM + px * 16;
    float s = 0.f;
    for (int y = 0; y < 16; ++y)
        for (int x = 0; x < 16; ++x)
            s += (float)src[y * HDIM + x];
    out[idx] = s * (1.f / 256.f);
}

// ---------------------------------------------------------------------------
// small scalar 3x3 conv (SAME) + BN + ReLU on tiny spatial grids (7x7 / 3x3)
// ---------------------------------------------------------------------------
__global__ void k_conv_small(const float* __restrict__ in, const float* __restrict__ w,
                             const float* __restrict__ g, const float* __restrict__ bb,
                             float* __restrict__ out, int Cin, int Cout, int Hd, int total) {
    int idx = blockIdx.x * blockDim.x + threadIdx.x;
    if (idx >= total) return;
    int sp = idx % (Hd * Hd);
    int t  = idx / (Hd * Hd);
    int oc = t % Cout;
    int b  = t / Cout;
    int oy = sp / Hd, ox = sp % Hd;
    float acc = 0.f;
    const float* wp = w + oc * Cin * 9;
    for (int ic = 0; ic < Cin; ++ic) {
        const float* ip = in + (b * Cin + ic) * Hd * Hd;
        #pragma unroll
        for (int ky = 0; ky < 3; ++ky) {
            int iy = oy + ky - 1;
            if ((unsigned)iy >= (unsigned)Hd) continue;
            #pragma unroll
            for (int kx = 0; kx < 3; ++kx) {
                int ix = ox + kx - 1;
                if ((unsigned)ix >= (unsigned)Hd) continue;
                acc += ip[iy * Hd + ix] * wp[ic * 9 + ky * 3 + kx];
            }
        }
    }
    float v = acc * (g[oc] * rsqrtf(1.f + BN_EPS)) + bb[oc];
    out[idx] = fmaxf(v, 0.f);
}

// ---------------------------------------------------------------------------
// AdaptiveAvgPool 7 -> 3 (window 3, stride 2)
// ---------------------------------------------------------------------------
__global__ void k_pool3(const float* __restrict__ in, float* __restrict__ out) {
    int idx = blockIdx.x * blockDim.x + threadIdx.x;    // 16*128*9
    if (idx >= BATCH * 128 * 9) return;
    int sp = idx % 9; int bc = idx / 9;
    int py = sp / 3, px = sp % 3;
    const float* src = in + bc * 49 + (py * 2) * 7 + px * 2;
    float s = 0.f;
    #pragma unroll
    for (int y = 0; y < 3; ++y)
        #pragma unroll
        for (int x = 0; x < 3; ++x)
            s += src[y * 7 + x];
    out[idx] = s * (1.f / 9.f);
}

// ---------------------------------------------------------------------------
// FC: feats[b,j] = sum_i h5[b,i] * w_fc[j,i]   (i<576, j<128)
// ---------------------------------------------------------------------------
__global__ void k_fc(const float* __restrict__ h, const float* __restrict__ wfc,
                     float* __restrict__ feats) {
    int idx = blockIdx.x * blockDim.x + threadIdx.x;    // 16*128
    if (idx >= BATCH * 128) return;
    int j = idx % 128, b = idx / 128;
    const float* hp = h + b * 576;
    const float* wp = wfc + j * 576;
    float acc = 0.f;
    for (int i = 0; i < 576; ++i) acc += hp[i] * wp[i];
    feats[idx] = acc;
}

// ---------------------------------------------------------------------------
// Bivariate normal mixture attention. One block per (b, out). 56x56 maps.
// ---------------------------------------------------------------------------
__global__ __launch_bounds__(256)
void k_attention(const float* __restrict__ feats, const float* __restrict__ mix_w,
                 float* __restrict__ out) {
    const int bo  = blockIdx.x;           // b*8 + o
    const int o   = bo & 7;
    const int tid = threadIdx.x;
    const int NPIX = 56 * 56;             // 3136

    __shared__ float att[4][56 * 56];
    __shared__ float rbuf[4][256];

    float mx[4], my[4], rr[4], rho[4], den[4];
    #pragma unroll
    for (int g = 0; g < 4; ++g) {
        const float* f = feats + bo * 16 + g * 4;   // feats[b, o*16 + g*4 + c]
        float f0 = f[0], f1 = f[1], f2 = f[2], f3 = f[3];
        float s0 = 1.f / (1.f + __expf(-f0));
        float s1 = 1.f / (1.f + __expf(-f1));
        float s2 = 1.f / (1.f + __expf(-f2));
        float s3 = 1.f / (1.f + __expf(-f3));
        mx[g]  = s0 * 55.f;
        my[g]  = s1 * 55.f;
        rr[g]  = __expf((s2 * 2.f - 1.f) * 1.0986123f);   // ln 3
        rho[g] = s3 * 1.6f - 0.8f;
        den[g] = -0.5f / (1.f - rho[g] * rho[g]) / 3136.f; // sigma = 56
    }

    float part[4] = {0.f, 0.f, 0.f, 0.f};
    for (int p = tid; p < NPIX; p += 256) {
        float i = (float)(p / 56);
        float j = (float)(p % 56);
        #pragma unroll
        for (int g = 0; g < 4; ++g) {
            float dx = mx[g] - i, dy = my[g] - j;
            float quad = dx * dx * rr[g] + dy * dy / rr[g] - 2.f * rho[g] * dx * dy;
            float e = __expf(den[g] * quad);
            att[g][p] = e;
            part[g] += e;
        }
    }
    #pragma unroll
    for (int g = 0; g < 4; ++g) rbuf[g][tid] = part[g];
    __syncthreads();
    for (int s = 128; s > 0; s >>= 1) {
        if (tid < s)
            #pragma unroll
            for (int g = 0; g < 4; ++g) rbuf[g][tid] += rbuf[g][tid + s];
        __syncthreads();
    }

    float w0 = mix_w[o * 4 + 0], w1 = mix_w[o * 4 + 1];
    float w2 = mix_w[o * 4 + 2], w3 = mix_w[o * 4 + 3];
    float wm = fmaxf(fmaxf(w0, w1), fmaxf(w2, w3));
    float e0 = __expf(w0 - wm), e1 = __expf(w1 - wm);
    float e2 = __expf(w2 - wm), e3 = __expf(w3 - wm);
    float inv_es = 1.f / (e0 + e1 + e2 + e3);
    float mwv[4] = {e0 * inv_es, e1 * inv_es, e2 * inv_es, e3 * inv_es};
    float inv[4];
    #pragma unroll
    for (int g = 0; g < 4; ++g) inv[g] = mwv[g] / rbuf[g][0];

    for (int p = tid; p < NPIX; p += 256) {
        float v = 0.f;
        #pragma unroll
        for (int g = 0; g < 4; ++g) v += att[g][p] * inv[g];
        out[bo * NPIX + p] = v;
    }
}

// ---------------------------------------------------------------------------
extern "C" void kernel_launch(void* const* d_in, const int* in_sizes, int n_in,
                              void* d_out, int out_size, void* d_ws, size_t ws_size,
                              hipStream_t stream) {
    const float* x    = (const float*)d_in[0];
    const float* w1   = (const float*)d_in[1];
    const float* g1   = (const float*)d_in[2];
    const float* b1   = (const float*)d_in[3];
    const float* w2   = (const float*)d_in[4];
    const float* g2   = (const float*)d_in[5];
    const float* b2   = (const float*)d_in[6];
    const float* w3   = (const float*)d_in[7];
    const float* g3   = (const float*)d_in[8];
    const float* b3   = (const float*)d_in[9];
    const float* w4   = (const float*)d_in[10];
    const float* g4   = (const float*)d_in[11];
    const float* b4   = (const float*)d_in[12];
    const float* w5   = (const float*)d_in[13];
    const float* g5   = (const float*)d_in[14];
    const float* b5   = (const float*)d_in[15];
    const float* wfc  = (const float*)d_in[16];
    const float* mixw = (const float*)d_in[17];

    uint8_t* ws = (uint8_t*)d_ws;
    size_t off = 0;
    auto alloc = [&](size_t bytes) { void* p = ws + off; off += (bytes + 255) & ~(size_t)255; return p; };

    bf16*  w1b   = (bf16*) alloc((size_t)256 * 4608 * 2);
    bf16*  w2b   = (bf16*) alloc((size_t)256 * 2304 * 2);
    bf16*  h1    = (bf16*) alloc((size_t)BATCH * 256 * HW * 2);
    bf16*  h2    = (bf16*) alloc((size_t)BATCH * 256 * HW * 2);
    float* p1    = (float*)alloc((size_t)BATCH * 256 * 49 * 4);
    float* h3    = (float*)alloc((size_t)BATCH * 128 * 49 * 4);
    float* h4    = (float*)alloc((size_t)BATCH * 128 * 49 * 4);
    float* p2    = (float*)alloc((size_t)BATCH * 128 * 9 * 4);
    float* h5    = (float*)alloc((size_t)BATCH * 64 * 9 * 4);
    float* feats = (float*)alloc((size_t)BATCH * 128 * 4);

    // weight conversion (flat OIHW order == GEMM [oc][ic*9+ky*3+kx])
    k_cvt_bf16<<<(256 * 4608 + 255) / 256, 256, 0, stream>>>(w1, w1b, 256 * 4608);
    k_cvt_bf16<<<(256 * 2304 + 255) / 256, 256, 0, stream>>>(w2, w2b, 256 * 2304);

    // conv1: [16,512,112,112] f32 -> [16,256,112,112] bf16   (WMMA + TDM)
    dim3 cg(BATCH * HW / 128, 2);   // (1568, 2)
    k_conv3x3_bn_relu_wmma<512, true ><<<cg, 256, 0, stream>>>(x,  w1b, g1, b1, h1, 256);
    // conv2: bf16 -> bf16   (WMMA + TDM)
    k_conv3x3_bn_relu_wmma<256, false><<<cg, 256, 0, stream>>>(h1, w2b, g2, b2, h2, 256);

    // pool 112->7
    k_pool16<<<(BATCH * 256 * 49) / 256, 256, 0, stream>>>(h2, p1);
    // conv3: 256->128 on 7x7
    k_conv_small<<<(BATCH * 128 * 49 + 255) / 256, 256, 0, stream>>>(p1, w3, g3, b3, h3, 256, 128, 7, BATCH * 128 * 49);
    // conv4: 128->128 on 7x7
    k_conv_small<<<(BATCH * 128 * 49 + 255) / 256, 256, 0, stream>>>(h3, w4, g4, b4, h4, 128, 128, 7, BATCH * 128 * 49);
    // pool 7->3
    k_pool3<<<(BATCH * 128 * 9 + 255) / 256, 256, 0, stream>>>(h4, p2);
    // conv5: 128->64 on 3x3
    k_conv_small<<<(BATCH * 64 * 9 + 255) / 256, 256, 0, stream>>>(p2, w5, g5, b5, h5, 128, 64, 3, BATCH * 64 * 9);
    // FC 576 -> 128
    k_fc<<<(BATCH * 128 + 255) / 256, 256, 0, stream>>>(h5, wfc, feats);
    // attention maps
    k_attention<<<BATCH * 8, 256, 0, stream>>>(feats, mixw, (float*)d_out);

    (void)in_sizes; (void)n_in; (void)out_size; (void)ws_size;
}
